// DecoderRNN_32710470926559
// MI455X (gfx1250) — matrix operand order repaired and checked by reading, not compiled, not run on hardware
//
#include <hip/hip_runtime.h>
#include <hip/hip_bf16.h>
#include <math.h>
#include <stdint.h>

// ---------------- problem constants ----------------
#define SOS_TOK 1
#define BB   32          // batch
#define TT   64          // time steps
#define HH   1024        // hidden
#define VV   32000       // vocab
#define GG   (3*HH)      // 3072 gate width
#define BH   (BB*HH)     // 32768
#define TB   (TT*BB)     // 2048

typedef __attribute__((ext_vector_type(2)))  float  v2f;
typedef __attribute__((ext_vector_type(8)))  float  v8f;
typedef __attribute__((ext_vector_type(4)))  __bf16 v4bf;
typedef __attribute__((ext_vector_type(8)))  __bf16 v8bf;
typedef __attribute__((ext_vector_type(16))) __bf16 v16bf;

// ---------------- workspace layout (floats) ----------------
#define X_OFF    ((size_t)0)
#define GI_OFF   ((size_t)2097152)
#define HS_OFF   (GI_OFF + (size_t)6291456)
#define GH_OFF   (HS_OFF + (size_t)2097152)
#define RMAX_OFF (GH_OFF + (size_t)98304)
#define RLSE_OFF (RMAX_OFF + (size_t)2048)

// ============================================================
// CDNA5 async global->LDS copy (16B per lane, ASYNCcnt tracked)
// Flat LDS pointer low 32 bits == wave-relative DS byte offset.
// ============================================================
__device__ __forceinline__ void async_g2l_b128(const void* gsrc, void* ldst) {
    uint32_t loff = (uint32_t)(uintptr_t)ldst;
    uint64_t gadr = (uint64_t)(uintptr_t)gsrc;
    asm volatile("global_load_async_to_lds_b128 %0, %1, off"
                 :: "v"(loff), "v"(gadr) : "memory");
}
__device__ __forceinline__ void wait_async0() {
    asm volatile("s_wait_asynccnt 0x0" ::: "memory");
}

// ============================================================
// Kernel 1: X[row] = relu(embedding[token(t,b)]), row = t*B + b
// ============================================================
__global__ __launch_bounds__(256)
void k_embed_relu(const int* __restrict__ target,      // [B, T]
                  const float* __restrict__ emb,       // [V, H]
                  float* __restrict__ X) {             // [T*B, H]
    const int row = blockIdx.x;
    const int t = row >> 5;
    const int b = row & 31;
    const int tok = (t == 0) ? SOS_TOK : target[b * TT + (t - 1)];
    const float* e = emb + (size_t)tok * HH;
    float* x = X + (size_t)row * HH;
    for (int c = threadIdx.x; c < HH; c += 256)
        x[c] = fmaxf(e[c], 0.0f);
}

// ============================================================
// Exact fp32 WMMA GEMM: C[M,N] = A[M,K] @ W[N,K]^T + bias[N]
//   used for the precision-critical GRU GEMMs (Gi, Gh).
//   256 thr = 8 waves; block tile 32x64; KT=32; async LDS staging.
// ============================================================
__global__ __launch_bounds__(256)
void k_gemm_wmma_f32(const float* __restrict__ A,
                     const float* __restrict__ W,
                     const float* __restrict__ bias,
                     float* __restrict__ C,
                     int M, int N, int K) {
    __shared__ __align__(16) float As[32][36];   // stride 36: 16B-aligned rows, no bank dup
    __shared__ __align__(16) float Bs[64][36];

    const int tid  = threadIdx.x;
    const int lane = tid & 31;
    const int wave = tid >> 5;
    const int l16  = lane & 15;
    const int koff = (lane >> 4) << 1;           // K sub-pair select: 0 or 2
    const int wm   = (wave >> 2) << 4;           // 0 / 16
    const int wn   = (wave & 3) << 4;            // 0 / 16 / 32 / 48

    const int n0g = blockIdx.x * 64;
    const int m0g = blockIdx.y * 32;

    // staging coords: one b128 per lane per pass
    const int sr = tid >> 3;                     // 0..31
    const int sc = (tid & 7) << 2;               // 0,4,..,28

    v8f acc = {0.f, 0.f, 0.f, 0.f, 0.f, 0.f, 0.f, 0.f};

    for (int k0 = 0; k0 < K; k0 += 32) {
        // A tile 32x32 floats: 256 lanes x 16B
        async_g2l_b128(A + (size_t)(m0g + sr) * K + k0 + sc, &As[sr][sc]);
        // W tile 64x32 floats: 2 passes
        async_g2l_b128(W + (size_t)(n0g + sr) * K + k0 + sc, &Bs[sr][sc]);
        async_g2l_b128(W + (size_t)(n0g + 32 + sr) * K + k0 + sc, &Bs[32 + sr][sc]);
        wait_async0();
        __syncthreads();

        #pragma unroll
        for (int kk = 0; kk < 32; kk += 4) {
            v2f a = *(const v2f*)&As[wm + l16][kk + koff];
            v2f b = *(const v2f*)&Bs[wn + l16][kk + koff];
            acc = __builtin_amdgcn_wmma_f32_16x16x4_f32(
                false, a, false, b, (short)0, acc, false, false);
        }
        __syncthreads();
    }

    const int rbase = m0g + wm + ((lane >> 4) << 3);
    const int gc    = n0g + wn + l16;
    const float bv  = bias[gc];
    #pragma unroll
    for (int j = 0; j < 8; ++j)
        C[(size_t)(rbase + j) * N + gc] = acc[j] + bv;
}

// ============================================================
// bf16 split-precision WMMA GEMM (near-fp32 accuracy):
//   x = hi + lo (bf16 each); x*w ~= hi*whi + hi*wlo + lo*whi
//   3 x v_wmma_f32_16x16x32_bf16 per 32-wide K tile.
//   PERMUTE: logical out-row r=(b*T+t) reads A row (t*B+b).
// ============================================================
struct bfpair { __bf16 hi; __bf16 lo; };
__device__ __forceinline__ bfpair bf16_split(float x) {
    bfpair p;
    p.hi = (__bf16)x;
    p.lo = (__bf16)(x - (float)p.hi);
    return p;
}

template <bool PERMUTE>
__global__ __launch_bounds__(256)
void k_gemm_wmma_bf16x(const float* __restrict__ A,
                       const float* __restrict__ W,
                       const float* __restrict__ bias,
                       float* __restrict__ C,
                       int M, int N, int K) {
    // stride 40 bf16 = 80B rows -> every 8-elem (16B) fragment read aligned
    __shared__ __align__(16) __bf16 As_hi[32][40];
    __shared__ __align__(16) __bf16 As_lo[32][40];
    __shared__ __align__(16) __bf16 Bs_hi[64][40];
    __shared__ __align__(16) __bf16 Bs_lo[64][40];

    const int tid  = threadIdx.x;
    const int lane = tid & 31;
    const int wave = tid >> 5;
    const int l16  = lane & 15;
    const int half = lane >> 4;
    const int k0h  = half << 3;                  // A frag K base: 0 or 8
    const int k1   = half << 4;                  // 0 or 16; unused (B frags also use k0h)
    const int wm   = (wave >> 2) << 4;
    const int wn   = (wave & 3) << 4;

    const int n0g = blockIdx.x * 64;
    const int m0g = blockIdx.y * 32;

    const int sr = tid >> 3;                     // 0..31
    const int sc = (tid & 7) << 2;               // 0,4,..,28 (float4 per lane)

    v8f acc = {0.f, 0.f, 0.f, 0.f, 0.f, 0.f, 0.f, 0.f};

    for (int k0 = 0; k0 < K; k0 += 32) {
        // ---- stage + split A tile (32x32 floats, 1 float4/lane) ----
        {
            const int gr = m0g + sr;
            const int ar = PERMUTE ? ((gr & (TT - 1)) * BB + (gr >> 6)) : gr;
            const float4 v = *(const float4*)(A + (size_t)ar * K + k0 + sc);
            const bfpair p0 = bf16_split(v.x), p1 = bf16_split(v.y);
            const bfpair p2 = bf16_split(v.z), p3 = bf16_split(v.w);
            v4bf h, l;
            h[0] = p0.hi; h[1] = p1.hi; h[2] = p2.hi; h[3] = p3.hi;
            l[0] = p0.lo; l[1] = p1.lo; l[2] = p2.lo; l[3] = p3.lo;
            *(v4bf*)&As_hi[sr][sc] = h;
            *(v4bf*)&As_lo[sr][sc] = l;
        }
        // ---- stage + split W tile (64x32 floats, 2 passes) ----
        #pragma unroll
        for (int q = 0; q < 2; ++q) {
            const int r = q * 32 + sr;
            const float4 v = *(const float4*)(W + (size_t)(n0g + r) * K + k0 + sc);
            const bfpair p0 = bf16_split(v.x), p1 = bf16_split(v.y);
            const bfpair p2 = bf16_split(v.z), p3 = bf16_split(v.w);
            v4bf h, l;
            h[0] = p0.hi; h[1] = p1.hi; h[2] = p2.hi; h[3] = p3.hi;
            l[0] = p0.lo; l[1] = p1.lo; l[2] = p2.lo; l[3] = p3.lo;
            *(v4bf*)&Bs_hi[r][sc] = h;
            *(v4bf*)&Bs_lo[r][sc] = l;
        }
        __syncthreads();

        // ---- fragments per ISA 16-bit layouts ----
        const int am = wm + l16;
        const int bn = wn + l16;
        v8bf a0 = *(const v8bf*)&As_hi[am][k0h];
        v8bf a1 = *(const v8bf*)&As_hi[am][k0h + 16];
        v16bf ah = __builtin_shufflevector(a0, a1, 0,1,2,3,4,5,6,7,8,9,10,11,12,13,14,15);
        a0 = *(const v8bf*)&As_lo[am][k0h];
        a1 = *(const v8bf*)&As_lo[am][k0h + 16];
        v16bf al = __builtin_shufflevector(a0, a1, 0,1,2,3,4,5,6,7,8,9,10,11,12,13,14,15);
        v8bf b0 = *(const v8bf*)&Bs_hi[bn][k0h];
        v8bf b1 = *(const v8bf*)&Bs_hi[bn][k0h + 16];
        v16bf bh = __builtin_shufflevector(b0, b1, 0,1,2,3,4,5,6,7,8,9,10,11,12,13,14,15);
        b0 = *(const v8bf*)&Bs_lo[bn][k0h];
        b1 = *(const v8bf*)&Bs_lo[bn][k0h + 16];
        v16bf bl = __builtin_shufflevector(b0, b1, 0,1,2,3,4,5,6,7,8,9,10,11,12,13,14,15);

        acc = __builtin_amdgcn_wmma_f32_16x16x32_bf16(
            false, ah, false, bh, (short)0, acc, false, false);
        acc = __builtin_amdgcn_wmma_f32_16x16x32_bf16(
            false, ah, false, bl, (short)0, acc, false, false);
        acc = __builtin_amdgcn_wmma_f32_16x16x32_bf16(
            false, al, false, bh, (short)0, acc, false, false);
        __syncthreads();
    }

    const int rbase = m0g + wm + ((lane >> 4) << 3);
    const int gc    = n0g + wn + l16;
    const float bv  = bias[gc];
    #pragma unroll
    for (int j = 0; j < 8; ++j)
        C[(size_t)(rbase + j) * N + gc] = acc[j] + bv;
}

// ============================================================
// fused GRU gates, one step
// ============================================================
__device__ __forceinline__ float sigmoidf_(float x) {
    return 1.0f / (1.0f + __expf(-x));
}

__global__ __launch_bounds__(256)
void k_gru_gate(const float* __restrict__ Gi_t,   // [B, 3H]
                const float* __restrict__ Gh,     // [B, 3H]
                const float* __restrict__ h_prev, // [B, H]
                float* __restrict__ h_out) {      // [B, H]
    const int i = blockIdx.x * 256 + threadIdx.x;
    const int b = i >> 10;
    const int j = i & (HH - 1);
    const float* gi = Gi_t + (size_t)b * GG;
    const float* gh = Gh   + (size_t)b * GG;
    const float r = sigmoidf_(gi[j]      + gh[j]);
    const float z = sigmoidf_(gi[HH + j] + gh[HH + j]);
    const float n = tanhf(gi[2 * HH + j] + r * gh[2 * HH + j]);
    h_out[i] = (1.0f - z) * n + z * h_prev[i];
}

// ============================================================
// per-row max and log-sum-exp over V (one block per row)
// ============================================================
__global__ __launch_bounds__(256)
void k_row_stats(const float* __restrict__ logits,
                 float* __restrict__ rowmax,
                 float* __restrict__ rowlse) {
    __shared__ float sm[256];
    const int r = blockIdx.x;
    const int tid = threadIdx.x;
    const float* p = logits + (size_t)r * VV;

    float m = -3.402823466e38f;
    for (int i = tid; i < VV; i += 256) m = fmaxf(m, p[i]);
    sm[tid] = m; __syncthreads();
    for (int s = 128; s > 0; s >>= 1) {
        if (tid < s) sm[tid] = fmaxf(sm[tid], sm[tid + s]);
        __syncthreads();
    }
    m = sm[0]; __syncthreads();

    float sum = 0.0f;
    for (int i = tid; i < VV; i += 256) sum += __expf(p[i] - m);
    sm[tid] = sum; __syncthreads();
    for (int s = 128; s > 0; s >>= 1) {
        if (tid < s) sm[tid] += sm[tid + s];
        __syncthreads();
    }
    if (tid == 0) { rowmax[r] = m; rowlse[r] = logf(sm[0]); }
}

// ============================================================
// in-place log-softmax normalize
// ============================================================
__global__ __launch_bounds__(256)
void k_normalize(float* __restrict__ out,
                 const float* __restrict__ rowmax,
                 const float* __restrict__ rowlse) {
    const int row = blockIdx.y;
    const int col = blockIdx.x * 256 + threadIdx.x;
    const size_t idx = (size_t)row * VV + col;
    out[idx] = out[idx] - rowmax[row] - rowlse[row];
}

// ============================================================
// copy final hidden state to output tail
// ============================================================
__global__ __launch_bounds__(256)
void k_copy_hidden(const float* __restrict__ h_last, float* __restrict__ dst) {
    const int i = blockIdx.x * 256 + threadIdx.x;
    dst[i] = h_last[i];
}

// ============================================================
// host-side orchestration
// ============================================================
extern "C" void kernel_launch(void* const* d_in, const int* in_sizes, int n_in,
                              void* d_out, int out_size, void* d_ws, size_t ws_size,
                              hipStream_t stream) {
    const float* enc_hidden = (const float*)d_in[1];   // [1,B,H]
    const int*   target     = (const int*)  d_in[2];   // [B,T]
    const float* embedding  = (const float*)d_in[3];   // [V,H]
    const float* w_ih       = (const float*)d_in[4];   // [3H,H]
    const float* w_hh       = (const float*)d_in[5];   // [3H,H]
    const float* b_ih       = (const float*)d_in[6];   // [3H]
    const float* b_hh       = (const float*)d_in[7];   // [3H]
    const float* w_out      = (const float*)d_in[8];   // [V,H]
    const float* b_out      = (const float*)d_in[9];   // [V]

    float* ws = (float*)d_ws;
    float* X      = ws + X_OFF;
    float* Gi     = ws + GI_OFF;
    float* hs     = ws + HS_OFF;
    float* Gh     = ws + GH_OFF;
    float* rowmax = ws + RMAX_OFF;
    float* rowlse = ws + RLSE_OFF;

    float* log_probs = (float*)d_out;                  // [B,T,V]
    float* h_out     = log_probs + (size_t)BB * TT * VV;

    // 1) embedding gather + relu for all T*B rows
    k_embed_relu<<<TB, 256, 0, stream>>>(target, embedding, X);

    // 2) Gi = X @ w_ih^T + b_ih   (M=2048, N=3072, K=1024) — exact fp32 WMMA
    k_gemm_wmma_f32<<<dim3(GG / 64, TB / 32), 256, 0, stream>>>(
        X, w_ih, b_ih, Gi, TB, GG, HH);

    // 3) sequential GRU recurrence (w_hh L2-resident across steps)
    for (int t = 0; t < TT; ++t) {
        const float* h_prev = (t == 0) ? enc_hidden : (hs + (size_t)(t - 1) * BH);
        k_gemm_wmma_f32<<<dim3(GG / 64, 1), 256, 0, stream>>>(
            h_prev, w_hh, b_hh, Gh, BB, GG, HH);
        k_gru_gate<<<BH / 256, 256, 0, stream>>>(
            Gi + (size_t)t * BB * GG, Gh, h_prev, hs + (size_t)t * BH);
    }

    // 4) logits[b*T+t] = hs[t*B+b] @ w_out^T + b_out  (M=2048, N=32000, K=1024)
    //    dominant GEMM -> bf16 split WMMA (3x 16x16x32_bf16 per K tile)
    k_gemm_wmma_bf16x<true><<<dim3(VV / 64, TB / 32), 256, 0, stream>>>(
        hs, w_out, b_out, log_probs, TB, VV, HH);

    // 5) per-row softmax statistics
    k_row_stats<<<TB, 256, 0, stream>>>(log_probs, rowmax, rowlse);

    // 6) in-place log-softmax
    k_normalize<<<dim3(VV / 256, TB), 256, 0, stream>>>(log_probs, rowmax, rowlse);

    // 7) final hidden state
    k_copy_hidden<<<BH / 256, 256, 0, stream>>>(hs + (size_t)(TT - 1) * BH, h_out);
}
